// GraphNetBlock_17008070492485
// MI455X (gfx1250) — matrix operand order repaired and checked by reading, not compiled, not run on hardware
//
#include <hip/hip_runtime.h>
#include <hip/hip_bf16.h>

typedef __attribute__((ext_vector_type(16))) __bf16 bf16x16;
typedef __attribute__((ext_vector_type(8)))  float   f32x8;

#define HDIM 128
#define LDK  136          // padded LDS row stride (bf16 elems) -> breaks 64-bank conflicts
#define LNEPS 1e-5f

union ABits {
  bf16x16 v;
  uint4   q[2];
  __bf16  e[16];
};

// B fragment: 32x16 bf16, lane(half,n): elems 0..7 = K0+half*8+i, elems 8..15 = K0+16+half*8+i
// -> two 8-bf16 (16B) chunks from transposed weights Wt[n][k]
static __device__ __forceinline__ bf16x16 load_bfrag(const __bf16* __restrict__ Wt,
                                                     int ldk, int col, int K0, int half) {
  ABits t;
  const uint4* p = reinterpret_cast<const uint4*>(Wt + (size_t)col * ldk + (K0 + half * 8));
  t.q[0] = p[0];
  t.q[1] = p[2];
  return t.v;
}

// A fragment: 16x32 bf16, lane(half,m): elems 0..7 = K0+half*8+i, elems 8..15 = K0+16+half*8+i
static __device__ __forceinline__ bf16x16 load_afrag_lds(const __bf16* act, int m, int K0, int half) {
  ABits t;
  t.q[0] = *reinterpret_cast<const uint4*>(act + m * LDK + K0 + half * 8);
  t.q[1] = *reinterpret_cast<const uint4*>(act + m * LDK + K0 + 16 + half * 8);
  return t.v;
}

static __device__ __forceinline__ void zero_acc(f32x8 acc[8]) {
  #pragma unroll
  for (int t = 0; t < 8; ++t)
    #pragma unroll
    for (int r = 0; r < 8; ++r)
      acc[t][r] = 0.0f;
}

static __device__ __forceinline__ void gemm_lds128(const __bf16* act, const __bf16* __restrict__ Wt,
                                                   f32x8 acc[8], int half, int n, int m) {
  for (int ks = 0; ks < 4; ++ks) {
    const int K0 = ks * 32;
    bf16x16 a = load_afrag_lds(act, m, K0, half);
    #pragma unroll
    for (int t = 0; t < 8; ++t) {
      bf16x16 b = load_bfrag(Wt, HDIM, t * 16 + n, K0, half);
      acc[t] = __builtin_amdgcn_wmma_f32_16x16x32_bf16(false, a, false, b,
                                                       (short)0, acc[t], false, false);
    }
  }
}

// x = relu(x + bias); x = (x - mean)/sqrt(var+eps) * g + beta   (row-wise over 128 cols)
// Row m = r + half*8 lives in accumulator reg r across one 16-lane half -> shfl_xor reduce.
static __device__ __forceinline__ void bias_relu_ln(f32x8 acc[8],
    const float* __restrict__ bias, const float* __restrict__ g,
    const float* __restrict__ beta, int n) {
  float sum[8], sq[8];
  #pragma unroll
  for (int r = 0; r < 8; ++r) { sum[r] = 0.f; sq[r] = 0.f; }
  #pragma unroll
  for (int t = 0; t < 8; ++t) {
    const float bv = bias[t * 16 + n];
    #pragma unroll
    for (int r = 0; r < 8; ++r) {
      float y = acc[t][r] + bv;
      y = fmaxf(y, 0.0f);
      acc[t][r] = y;
      sum[r] += y;
      sq[r]  += y * y;
    }
  }
  #pragma unroll
  for (int r = 0; r < 8; ++r) {
    #pragma unroll
    for (int msk = 1; msk < 16; msk <<= 1) {
      sum[r] += __shfl_xor(sum[r], msk, 32);
      sq[r]  += __shfl_xor(sq[r],  msk, 32);
    }
  }
  float mean[8], rstd[8];
  #pragma unroll
  for (int r = 0; r < 8; ++r) {
    mean[r] = sum[r] * (1.0f / HDIM);
    float var = sq[r] * (1.0f / HDIM) - mean[r] * mean[r];
    rstd[r] = rsqrtf(var + LNEPS);
  }
  #pragma unroll
  for (int t = 0; t < 8; ++t) {
    const float gv = g[t * 16 + n];
    const float bv = beta[t * 16 + n];
    #pragma unroll
    for (int r = 0; r < 8; ++r)
      acc[t][r] = (acc[t][r] - mean[r]) * rstd[r] * gv + bv;
  }
}

static __device__ __forceinline__ void store_act(__bf16* act, f32x8 acc[8], int half, int n) {
  #pragma unroll
  for (int t = 0; t < 8; ++t)
    #pragma unroll
    for (int r = 0; r < 8; ++r)
      act[(r + half * 8) * LDK + t * 16 + n] = (__bf16)acc[t][r];
}

static __device__ __forceinline__ void cvt8(ABits& af, int gq, const float* p) {
  const float4 f0 = *reinterpret_cast<const float4*>(p);
  const float4 f1 = *reinterpret_cast<const float4*>(p + 4);
  af.e[gq*8+0] = (__bf16)f0.x; af.e[gq*8+1] = (__bf16)f0.y;
  af.e[gq*8+2] = (__bf16)f0.z; af.e[gq*8+3] = (__bf16)f0.w;
  af.e[gq*8+4] = (__bf16)f1.x; af.e[gq*8+5] = (__bf16)f1.y;
  af.e[gq*8+6] = (__bf16)f1.z; af.e[gq*8+7] = (__bf16)f1.w;
}

// ---------------- Edge kernel: concat[edge_attr, x[src], x[dst]] -> MLP -> out + scatter ----
extern "C" __global__ __launch_bounds__(256)
void gnb_edge_kernel(const float* __restrict__ x, const float* __restrict__ ea,
                     const int* __restrict__ src, const int* __restrict__ dst,
                     const __bf16* __restrict__ Wt1, const __bf16* __restrict__ Wt2,
                     const __bf16* __restrict__ Wt3, const __bf16* __restrict__ Wt4,
                     const float* __restrict__ b1, const float* __restrict__ b2,
                     const float* __restrict__ b3, const float* __restrict__ b4,
                     const float* __restrict__ g1, const float* __restrict__ g2,
                     const float* __restrict__ g3,
                     const float* __restrict__ be1, const float* __restrict__ be2,
                     const float* __restrict__ be3,
                     float* __restrict__ agg, float* __restrict__ outE, int E)
{
  __shared__ __bf16 actS[8][16 * LDK];
  const int wave = threadIdx.x >> 5;
  const int lane = threadIdx.x & 31;
  const int half = lane >> 4;
  const int n = lane & 15;
  const int m = n;
  __bf16* act = actS[wave];
  const int base = blockIdx.x * 128 + wave * 16;

  const int eA = min(base + m, E - 1);
  const int sA = src[eA];
  const int dA = dst[eA];
  const float* seg0 = ea + (size_t)eA * HDIM;
  const float* seg1 = x + (size_t)sA * HDIM;
  const float* seg2 = x + (size_t)dA * HDIM;

  f32x8 acc[8];
  zero_acc(acc);
  // layer 1: K = 384 (gathered, f32->bf16 on the fly)
  for (int ks = 0; ks < 12; ++ks) {
    const int K0 = ks * 32;
    ABits af;
    #pragma unroll
    for (int gq = 0; gq < 2; ++gq) {
      const int kabs = K0 + gq * 16 + half * 8;
      const int segi = kabs >> 7;
      const float* p = (segi == 0 ? seg0 : (segi == 1 ? seg1 : seg2)) + (kabs & 127);
      cvt8(af, gq, p);
    }
    #pragma unroll
    for (int t = 0; t < 8; ++t) {
      bf16x16 b = load_bfrag(Wt1, 3 * HDIM, t * 16 + n, K0, half);
      acc[t] = __builtin_amdgcn_wmma_f32_16x16x32_bf16(false, af.v, false, b,
                                                       (short)0, acc[t], false, false);
    }
  }
  bias_relu_ln(acc, b1, g1, be1, n);
  store_act(act, acc, half, n);
  __syncthreads();

  zero_acc(acc);
  gemm_lds128(act, Wt2, acc, half, n, m);
  bias_relu_ln(acc, b2, g2, be2, n);
  __syncthreads();
  store_act(act, acc, half, n);
  __syncthreads();

  zero_acc(acc);
  gemm_lds128(act, Wt3, acc, half, n, m);
  bias_relu_ln(acc, b3, g3, be3, n);
  __syncthreads();
  store_act(act, acc, half, n);
  __syncthreads();

  zero_acc(acc);
  gemm_lds128(act, Wt4, acc, half, n, m);

  int rowO[8], dstO[8];
  #pragma unroll
  for (int r = 0; r < 8; ++r) {
    rowO[r] = base + r + half * 8;
    dstO[r] = dst[min(rowO[r], E - 1)];
  }
  #pragma unroll
  for (int t = 0; t < 8; ++t) {
    const float bv = b4[t * 16 + n];
    const int col = t * 16 + n;
    #pragma unroll
    for (int r = 0; r < 8; ++r) {
      if (rowO[r] < E) {
        const float v = acc[t][r] + bv;                 // edge_new
        const size_t off = (size_t)rowO[r] * HDIM + col;
        outE[off] = ea[off] + v;                        // residual
        atomicAdd(agg + (size_t)dstO[r] * HDIM + col, v); // scatter-add (no-return f32 atomic)
      }
    }
  }
}

// ---------------- Node kernel: concat[x, agg] -> MLP -> residual out ------------------------
extern "C" __global__ __launch_bounds__(256)
void gnb_node_kernel(const float* __restrict__ x, const float* __restrict__ agg,
                     const __bf16* __restrict__ Wt1, const __bf16* __restrict__ Wt2,
                     const __bf16* __restrict__ Wt3, const __bf16* __restrict__ Wt4,
                     const float* __restrict__ b1, const float* __restrict__ b2,
                     const float* __restrict__ b3, const float* __restrict__ b4,
                     const float* __restrict__ g1, const float* __restrict__ g2,
                     const float* __restrict__ g3,
                     const float* __restrict__ be1, const float* __restrict__ be2,
                     const float* __restrict__ be3,
                     float* __restrict__ outN, int Nn)
{
  __shared__ __bf16 actS[8][16 * LDK];
  const int wave = threadIdx.x >> 5;
  const int lane = threadIdx.x & 31;
  const int half = lane >> 4;
  const int n = lane & 15;
  const int m = n;
  __bf16* act = actS[wave];
  const int base = blockIdx.x * 128 + wave * 16;

  const int rA = min(base + m, Nn - 1);
  const float* seg0 = x + (size_t)rA * HDIM;
  const float* seg1 = agg + (size_t)rA * HDIM;

  f32x8 acc[8];
  zero_acc(acc);
  // layer 1: K = 256
  for (int ks = 0; ks < 8; ++ks) {
    const int K0 = ks * 32;
    ABits af;
    #pragma unroll
    for (int gq = 0; gq < 2; ++gq) {
      const int kabs = K0 + gq * 16 + half * 8;
      const float* p = ((kabs >> 7) ? seg1 : seg0) + (kabs & 127);
      cvt8(af, gq, p);
    }
    #pragma unroll
    for (int t = 0; t < 8; ++t) {
      bf16x16 b = load_bfrag(Wt1, 2 * HDIM, t * 16 + n, K0, half);
      acc[t] = __builtin_amdgcn_wmma_f32_16x16x32_bf16(false, af.v, false, b,
                                                       (short)0, acc[t], false, false);
    }
  }
  bias_relu_ln(acc, b1, g1, be1, n);
  store_act(act, acc, half, n);
  __syncthreads();

  zero_acc(acc);
  gemm_lds128(act, Wt2, acc, half, n, m);
  bias_relu_ln(acc, b2, g2, be2, n);
  __syncthreads();
  store_act(act, acc, half, n);
  __syncthreads();

  zero_acc(acc);
  gemm_lds128(act, Wt3, acc, half, n, m);
  bias_relu_ln(acc, b3, g3, be3, n);
  __syncthreads();
  store_act(act, acc, half, n);
  __syncthreads();

  zero_acc(acc);
  gemm_lds128(act, Wt4, acc, half, n, m);

  #pragma unroll
  for (int t = 0; t < 8; ++t) {
    const float bv = b4[t * 16 + n];
    const int col = t * 16 + n;
    #pragma unroll
    for (int r = 0; r < 8; ++r) {
      const int row = base + r + half * 8;
      if (row < Nn) {
        const size_t off = (size_t)row * HDIM + col;
        outN[off] = x[off] + acc[t][r] + bv;   // residual
      }
    }
  }
}

// ---------------- Weight prep: W[K][N] f32 -> Wt[N][K] bf16 --------------------------------
extern "C" __global__ void gnb_prep_wt(const float* __restrict__ W, __bf16* __restrict__ Wt,
                                       int K, int N) {
  const int idx = blockIdx.x * 256 + threadIdx.x;
  if (idx >= K * N) return;
  const int k = idx % K;
  const int nn = idx / K;
  Wt[(size_t)nn * K + k] = (__bf16)W[(size_t)k * N + nn];
}

extern "C" void kernel_launch(void* const* d_in, const int* in_sizes, int n_in,
                              void* d_out, int out_size, void* d_ws, size_t ws_size,
                              hipStream_t stream) {
  (void)n_in; (void)out_size; (void)ws_size;
  const int Hc = 128;
  const int Nn = in_sizes[0] / Hc;
  const int E  = in_sizes[1] / Hc;

  const float* x  = (const float*)d_in[0];
  const float* ea = (const float*)d_in[1];
  const int*  eidx = (const int*)d_in[2];
  const int* src = eidx;
  const int* dst = eidx + E;

  const float* eW[4] = { (const float*)d_in[3], (const float*)d_in[5],
                         (const float*)d_in[7], (const float*)d_in[9] };
  const float* eb[4] = { (const float*)d_in[4], (const float*)d_in[6],
                         (const float*)d_in[8], (const float*)d_in[10] };
  const float* eg[3] = { (const float*)d_in[11], (const float*)d_in[13], (const float*)d_in[15] };
  const float* ebe[3]= { (const float*)d_in[12], (const float*)d_in[14], (const float*)d_in[16] };
  const float* nW[4] = { (const float*)d_in[17], (const float*)d_in[19],
                         (const float*)d_in[21], (const float*)d_in[23] };
  const float* nb[4] = { (const float*)d_in[18], (const float*)d_in[20],
                         (const float*)d_in[22], (const float*)d_in[24] };
  const float* ng[3] = { (const float*)d_in[25], (const float*)d_in[27], (const float*)d_in[29] };
  const float* nbe[3]= { (const float*)d_in[26], (const float*)d_in[28], (const float*)d_in[30] };

  // workspace layout: transposed bf16 weights, then f32 agg buffer
  __bf16* wsb  = (__bf16*)d_ws;
  __bf16* eWt1 = wsb;
  __bf16* eWt2 = eWt1 + 128 * 384;
  __bf16* eWt3 = eWt2 + 128 * 128;
  __bf16* eWt4 = eWt3 + 128 * 128;
  __bf16* nWt1 = eWt4 + 128 * 128;
  __bf16* nWt2 = nWt1 + 128 * 256;
  __bf16* nWt3 = nWt2 + 128 * 128;
  __bf16* nWt4 = nWt3 + 128 * 128;
  size_t wElems = (size_t)(128 * 384 + 3 * 128 * 128 + 128 * 256 + 3 * 128 * 128);
  wElems = (wElems + 127) & ~(size_t)127;   // 256B align
  float* agg = (float*)(wsb + wElems);

  struct PJ { const float* W; __bf16* Wt; int K; int N; };
  const PJ jobs[8] = {
    { eW[0], eWt1, 384, 128 }, { eW[1], eWt2, 128, 128 },
    { eW[2], eWt3, 128, 128 }, { eW[3], eWt4, 128, 128 },
    { nW[0], nWt1, 256, 128 }, { nW[1], nWt2, 128, 128 },
    { nW[2], nWt3, 128, 128 }, { nW[3], nWt4, 128, 128 },
  };
  for (int j = 0; j < 8; ++j) {
    const int tot = jobs[j].K * jobs[j].N;
    gnb_prep_wt<<<(tot + 255) / 256, 256, 0, stream>>>(jobs[j].W, jobs[j].Wt,
                                                       jobs[j].K, jobs[j].N);
  }

  hipMemsetAsync(agg, 0, (size_t)Nn * Hc * sizeof(float), stream);

  float* outN = (float*)d_out;
  float* outE = outN + (size_t)Nn * Hc;

  gnb_edge_kernel<<<(E + 127) / 128, 256, 0, stream>>>(
      x, ea, src, dst, eWt1, eWt2, eWt3, eWt4,
      eb[0], eb[1], eb[2], eb[3], eg[0], eg[1], eg[2], ebe[0], ebe[1], ebe[2],
      agg, outE, E);

  gnb_node_kernel<<<(Nn + 127) / 128, 256, 0, stream>>>(
      x, agg, nWt1, nWt2, nWt3, nWt4,
      nb[0], nb[1], nb[2], nb[3], ng[0], ng[1], ng[2], nbe[0], nbe[1], nbe[2],
      outN, Nn);
}
